// SDPAAttention_32573031973657
// MI455X (gfx1250) — matrix operand order repaired and checked by reading, not compiled, not run on hardware
//
#include <hip/hip_runtime.h>

#define H_   12
#define D_   64
#define N_   576
#define C_   768
#define MTOT 13914   // 15*6 special rows + 72*192 cache rows
#define MPAD 13952   // round up to 64
#define MOLD 13338   // rows sourced from old cache/special; new keys at [MOLD, MTOT)
#define PARTS 8
#define PTILE 1792   // 28 tiles of 64 keys per partition (last part gets 22)
#define PSTRIDE 1056 // floats per partial: 16x64 O + 16 m + 16 l

typedef __attribute__((ext_vector_type(16))) __bf16 v16bf;
typedef __attribute__((ext_vector_type(8)))  __bf16 v8bf;
typedef __attribute__((ext_vector_type(8)))  float  v8f;

__device__ __forceinline__ unsigned short f32_bf16(float f) {
  unsigned int u = __float_as_uint(f);
  u += 0x7FFFu + ((u >> 16) & 1u);          // round-to-nearest-even
  return (unsigned short)(u >> 16);
}

// One wave computes a 16(M) x 64(N) f32 tile, K in steps of 32, bf16 inputs.
// Software-pipelined: fragment loads for step ks+1 are issued before the WMMAs
// of step ks, so the compiler can overlap global/L2 latency with matrix math.
// A-frag per lane: row=lane&15, 16 consecutive K (hi half: +16).
// B-frag per lane: col=lane&15 (+16*t), 16 consecutive K (hi half: +16).
template <int KSTEPS>
__device__ __forceinline__ void wave_tile_16x64(const unsigned short* __restrict__ A, int lda,
                                                const unsigned short* __restrict__ B, long ldb,
                                                v8f* acc) {
  const int lane = threadIdx.x & 31;
  const int r16  = lane & 15;
  const int hi   = lane >> 4;
  const unsigned short* ap = A + r16 * lda + hi * 16;
  const unsigned short* bp = B + (long)r16 * ldb + hi * 16;
  union AU { v16bf v; v8bf h[2]; };
  AU a[2];
  v16bf b[2][4];
  a[0].h[0] = *(const v8bf*)(ap);
  a[0].h[1] = *(const v8bf*)(ap + 8);
#pragma unroll
  for (int t = 0; t < 4; ++t) b[0][t] = *(const v16bf*)(bp + (long)t * 16 * ldb);
#pragma unroll
  for (int ks = 0; ks < KSTEPS; ++ks) {
    const int cur = ks & 1, nxt = cur ^ 1;
    if (ks + 1 < KSTEPS) {                      // issue next step's loads first
      const unsigned short* ap2 = ap + (ks + 1) * 32;
      const unsigned short* bp2 = bp + (ks + 1) * 32;
      a[nxt].h[0] = *(const v8bf*)(ap2);
      a[nxt].h[1] = *(const v8bf*)(ap2 + 8);
#pragma unroll
      for (int t = 0; t < 4; ++t) b[nxt][t] = *(const v16bf*)(bp2 + (long)t * 16 * ldb);
    }
#pragma unroll
    for (int t = 0; t < 4; ++t)
      acc[t] = __builtin_amdgcn_wmma_f32_16x16x32_bf16(false, a[cur].v, false, b[cur][t],
                                                       (short)0, acc[t], false, false);
  }
}

// Gather map for m in [0, MOLD): reproduces concat(special+new_sk, cache[:8]+cache[11:72]).
__device__ __forceinline__ float kv_gather(const float* __restrict__ cache,
                                           const float* __restrict__ special,
                                           int h, int m, int d) {
  if (m < 72) {            // original k_special (12 frames x 6)
    int s = m / 6, t = m - s * 6;
    return special[((h * 12 + s) * 6 + t) * 64 + d];
  }
  if (m < 90) {            // new_sk: evicted cache frames 8..10, first 6 tokens
    int j = m - 72; int q = j / 6; int f = 8 + q, t = j - q * 6;
    return cache[(((long)h * 72 + f) * 192 + t) * 64 + d];
  }
  if (m < MOLD) {          // kept cache frames: 0..7 then 11..71
    int j = m - 90; int cf = j / 192, t = j - cf * 192;
    int f = cf < 8 ? cf : cf + 3;
    return cache[(((long)h * 72 + f) * 192 + t) * 64 + d];
  }
  return 0.0f;             // new-key region + pad: zero now, GEMM overwrites [MOLD,MTOT)
}

__global__ void conv_bf16_kernel(const float* __restrict__ src, unsigned short* __restrict__ dst, int n) {
  int i = blockIdx.x * blockDim.x + threadIdx.x;
  if (i < n) dst[i] = f32_bf16(src[i]);
}

// KBF layout: (h, m, d) row-major bf16.  Flat index == thread index.
__global__ void gather_k_kernel(const float* __restrict__ k_cache, const float* __restrict__ k_special,
                                unsigned short* __restrict__ KBF) {
  long i = (long)blockIdx.x * blockDim.x + threadIdx.x;   // over H_*MPAD*64
  int d = (int)(i & 63);
  long hm = i >> 6;
  int m = (int)(hm % MPAD);
  int h = (int)(hm / MPAD);
  KBF[i] = f32_bf16(kv_gather(k_cache, k_special, h, m, d));
}

// VBFT layout: (h, d, m) bf16 — LDS-tiled transpose, coalesced both sides.
__global__ __launch_bounds__(256) void gather_vt_kernel(const float* __restrict__ v_cache,
                                                        const float* __restrict__ v_special,
                                                        unsigned short* __restrict__ VBFT) {
  __shared__ float tile[64][65];
  const int h = blockIdx.y;
  const int mbase = blockIdx.x * 64;
  const int tid = threadIdx.x;
#pragma unroll
  for (int i = 0; i < 16; ++i) {
    int lin = i * 256 + tid;
    int ml = lin >> 6, d = lin & 63;
    tile[ml][d] = kv_gather(v_cache, v_special, h, mbase + ml, d);
  }
  __syncthreads();
#pragma unroll
  for (int i = 0; i < 16; ++i) {
    int lin = i * 256 + tid;
    int dl = lin >> 6, ml = lin & 63;
    VBFT[((long)h * 64 + dl) * MPAD + mbase + ml] = f32_bf16(tile[ml][dl]);
  }
}

// QKV projection: 576x2304 = Xbf(576,768) @ Wbf(2304,768)^T, epilogue routes q/k/v.
__global__ __launch_bounds__(32) void qkv_gemm_kernel(const unsigned short* __restrict__ Xbf,
                                                      const unsigned short* __restrict__ Wbf,
                                                      const float* __restrict__ qkv_b,
                                                      unsigned short* __restrict__ QBF,
                                                      unsigned short* __restrict__ KBF,
                                                      unsigned short* __restrict__ VBFT) {
  const int nt = blockIdx.x;          // 36 row tiles of 16
  const int ot = blockIdx.y;          // 36 col tiles of 64 (covers 2304)
  v8f acc[4];
#pragma unroll
  for (int t = 0; t < 4; ++t) acc[t] = 0.0f;
  wave_tile_16x64<C_ / 32>(Xbf + nt * 16 * C_, C_, Wbf + (long)ot * 64 * C_, C_, acc);
  const int lane = threadIdx.x & 31;
  const int col = lane & 15, rowoff = (lane >> 4) * 8;
#pragma unroll
  for (int t = 0; t < 4; ++t) {
    int o = ot * 64 + t * 16 + col;
    float bias = qkv_b[o];
    int which = o / 768;
    int rem = o - which * 768;
    int h = rem >> 6, d = rem & 63;
#pragma unroll
    for (int r = 0; r < 8; ++r) {
      int n = nt * 16 + r + rowoff;
      float v = acc[t][r] + bias;
      if (which == 0) {
        QBF[((long)h * N_ + n) * 64 + d] = f32_bf16(v * 0.125f);   // fold 1/sqrt(D)
      } else if (which == 1) {
        KBF[((long)h * MPAD + MOLD + n) * 64 + d] = f32_bf16(v);
      } else {
        VBFT[((long)h * 64 + d) * MPAD + MOLD + n] = f32_bf16(v);
      }
    }
  }
}

// Flash-decoding attention partial: one wave per (head, 16 q rows, M-partition).
// Emits unnormalized O plus per-row (m, l) so a second pass can merge partitions.
__global__ __launch_bounds__(32) void attn_part_kernel(const unsigned short* __restrict__ QBF,
                                                       const unsigned short* __restrict__ KBF,
                                                       const unsigned short* __restrict__ VBFT,
                                                       float* __restrict__ PART) {
  __shared__ unsigned short Plds[16 * 72];   // P tile, row stride 72 to dodge bank conflicts
  const int qt = blockIdx.x;                 // 36
  const int h  = blockIdx.y;                 // 12
  const int p  = blockIdx.z;                 // PARTS
  const int qbase = qt * 16;
  const int lane = threadIdx.x & 31;
  const int r16 = lane & 15, hi = lane >> 4;
  const int col = r16, rowoff = hi * 8;
  const unsigned short* Aq = QBF + ((long)h * N_ + qbase) * 64;
  const unsigned short* Kh = KBF + (long)h * MPAD * 64;
  const unsigned short* Vh = VBFT + (long)h * 64 * MPAD;
  const int mstart = p * PTILE;
  const int mend = (mstart + PTILE < MPAD) ? (mstart + PTILE) : MPAD;

  float mrow[8], lrow[8];
  v8f O[4];
#pragma unroll
  for (int r = 0; r < 8; ++r) { mrow[r] = -3.0e38f; lrow[r] = 0.0f; }
#pragma unroll
  for (int t = 0; t < 4; ++t) O[t] = 0.0f;

  for (int mb = mstart; mb < mend; mb += 64) {
    if (mb + 64 < mend) {
      __builtin_prefetch(Kh + (long)(mb + 64) * 64, 0, 0);
      __builtin_prefetch(Vh + mb + 64, 0, 0);
    }
    // scores S = Q K^T (Q pre-scaled), 16 q x 64 keys
    v8f s[4];
#pragma unroll
    for (int t = 0; t < 4; ++t) s[t] = 0.0f;
    wave_tile_16x64<2>(Aq, 64, Kh + (long)mb * 64, 64, s);

    // Issue the V fragment loads NOW: they don't depend on softmax, so the
    // whole f32 softmax VALU burst below hides their L2 latency.
    v16bf vf[2][4];
#pragma unroll
    for (int ks = 0; ks < 2; ++ks)
#pragma unroll
      for (int t = 0; t < 4; ++t)
        vf[ks][t] = *(const v16bf*)(Vh + (long)(r16 + 16 * t) * MPAD + mb + hi * 16 + ks * 32);

    if (mb + 64 > MTOT) {                      // mask pad keys (wave-uniform branch)
#pragma unroll
      for (int t = 0; t < 4; ++t) {
        int j = mb + t * 16 + col;
        if (j >= MTOT) {
#pragma unroll
          for (int r = 0; r < 8; ++r) s[t][r] = -3.0e38f;
        }
      }
    }
    // online softmax: C/D layout puts row r (lanes 0-15) / r+8 (lanes 16-31) in slot r,
    // so xor-shuffles over masks {1,2,4,8} reduce each row within its lane half.
    float mt[8];
#pragma unroll
    for (int r = 0; r < 8; ++r)
      mt[r] = fmaxf(fmaxf(s[0][r], s[1][r]), fmaxf(s[2][r], s[3][r]));
#pragma unroll
    for (int k = 1; k < 16; k <<= 1)
#pragma unroll
      for (int r = 0; r < 8; ++r) mt[r] = fmaxf(mt[r], __shfl_xor(mt[r], k, 32));
    float al[8], rs[8];
#pragma unroll
    for (int r = 0; r < 8; ++r) {
      float mn = fmaxf(mrow[r], mt[r]);
      al[r] = __expf(mrow[r] - mn);
      mrow[r] = mn;
      rs[r] = 0.0f;
    }
#pragma unroll
    for (int t = 0; t < 4; ++t)
#pragma unroll
      for (int r = 0; r < 8; ++r) {
        float pv = __expf(s[t][r] - mrow[r]);
        s[t][r] = pv;
        rs[r] += pv;
      }
#pragma unroll
    for (int k = 1; k < 16; k <<= 1)
#pragma unroll
      for (int r = 0; r < 8; ++r) rs[r] += __shfl_xor(rs[r], k, 32);
#pragma unroll
    for (int r = 0; r < 8; ++r) lrow[r] = lrow[r] * al[r] + rs[r];
#pragma unroll
    for (int t = 0; t < 4; ++t)
#pragma unroll
      for (int r = 0; r < 8; ++r) O[t][r] *= al[r];
    // stage P to LDS (bf16), reload as A-fragments (DS ops are in-order in-wave)
#pragma unroll
    for (int t = 0; t < 4; ++t)
#pragma unroll
      for (int r = 0; r < 8; ++r)
        Plds[(r + rowoff) * 72 + t * 16 + col] = f32_bf16(s[t][r]);
    union AU { v16bf v; v8bf hh[2]; } aP[2];
#pragma unroll
    for (int ks = 0; ks < 2; ++ks) {
      const unsigned short* pp = Plds + r16 * 72 + hi * 16 + ks * 32;
      aP[ks].hh[0] = *(const v8bf*)(pp);
      aP[ks].hh[1] = *(const v8bf*)(pp + 8);
    }
#pragma unroll
    for (int ks = 0; ks < 2; ++ks)
#pragma unroll
      for (int t = 0; t < 4; ++t)
        O[t] = __builtin_amdgcn_wmma_f32_16x16x32_bf16(false, aP[ks].v, false, vf[ks][t],
                                                       (short)0, O[t], false, false);
  }
  // write partial: [0,1024) O row-major, [1024,1040) m, [1040,1056) l
  float* pb = PART + ((long)((h * 36 + qt) * PARTS) + p) * PSTRIDE;
#pragma unroll
  for (int t = 0; t < 4; ++t)
#pragma unroll
    for (int r = 0; r < 8; ++r)
      pb[(r + rowoff) * 64 + t * 16 + col] = O[t][r];
#pragma unroll
  for (int r = 0; r < 8; ++r) {
    if (col == r) {
      pb[1024 + r + rowoff] = mrow[r];
      pb[1040 + r + rowoff] = lrow[r];
    }
  }
}

// Merge the PARTS partials with log-sum-exp weights; emit bf16 concat-head activation.
__global__ __launch_bounds__(128) void attn_reduce_kernel(const float* __restrict__ PART,
                                                          unsigned short* __restrict__ ATTNBF) {
  __shared__ float sm[16], sl[16];
  const int qt = blockIdx.x, h = blockIdx.y;
  const float* base = PART + (long)((h * 36 + qt) * PARTS) * PSTRIDE;
  const int tid = threadIdx.x;
  if (tid < 16) {
    float mmax = -3.0e38f;
#pragma unroll
    for (int p = 0; p < PARTS; ++p)
      mmax = fmaxf(mmax, base[p * PSTRIDE + 1024 + tid]);
    float ls = 0.0f;
#pragma unroll
    for (int p = 0; p < PARTS; ++p)
      ls += base[p * PSTRIDE + 1040 + tid] * __expf(base[p * PSTRIDE + 1024 + tid] - mmax);
    sm[tid] = mmax;
    sl[tid] = ls;
  }
  __syncthreads();
#pragma unroll
  for (int i = tid; i < 1024; i += 128) {
    int row = i >> 6, d = i & 63;
    float acc = 0.0f;
#pragma unroll
    for (int p = 0; p < PARTS; ++p)
      acc += base[p * PSTRIDE + i] * __expf(base[p * PSTRIDE + 1024 + row] - sm[row]);
    int n = qt * 16 + row;
    ATTNBF[(long)n * C_ + h * 64 + d] = f32_bf16(acc / sl[row]);
  }
}

// Output projection: out(576,768) = attn @ proj_w^T + proj_b, f32 result.
__global__ __launch_bounds__(32) void proj_gemm_kernel(const unsigned short* __restrict__ Abf,
                                                       const unsigned short* __restrict__ Wbf,
                                                       const float* __restrict__ proj_b,
                                                       float* __restrict__ out) {
  const int nt = blockIdx.x;          // 36
  const int ot = blockIdx.y;          // 12
  v8f acc[4];
#pragma unroll
  for (int t = 0; t < 4; ++t) acc[t] = 0.0f;
  wave_tile_16x64<C_ / 32>(Abf + nt * 16 * C_, C_, Wbf + (long)ot * 64 * C_, C_, acc);
  const int lane = threadIdx.x & 31;
  const int col = lane & 15, rowoff = (lane >> 4) * 8;
#pragma unroll
  for (int t = 0; t < 4; ++t) {
    int o = ot * 64 + t * 16 + col;
    float bias = proj_b[o];
#pragma unroll
    for (int r = 0; r < 8; ++r) {
      int n = nt * 16 + r + rowoff;
      out[(long)n * C_ + o] = acc[t][r] + bias;
    }
  }
}

extern "C" void kernel_launch(void* const* d_in, const int* in_sizes, int n_in,
                              void* d_out, int out_size, void* d_ws, size_t ws_size,
                              hipStream_t stream) {
  const float* x         = (const float*)d_in[0];   // (1,576,768)
  const float* qkv_w     = (const float*)d_in[1];   // (2304,768)
  const float* qkv_b     = (const float*)d_in[2];   // (2304)
  const float* proj_w    = (const float*)d_in[3];   // (768,768)
  const float* proj_b    = (const float*)d_in[4];   // (768)
  const float* k_cache   = (const float*)d_in[5];   // (1,12,72,192,64)
  const float* v_cache   = (const float*)d_in[6];
  const float* k_special = (const float*)d_in[7];   // (1,12,12,6,64)
  const float* v_special = (const float*)d_in[8];
  float* out = (float*)d_out;

  // workspace carve-up (256B aligned)
  char* ws = (char*)d_ws;
  size_t off = 0;
  auto carve = [&](size_t bytes) { void* ptr = ws + off; off += (bytes + 255) & ~(size_t)255; return ptr; };
  unsigned short* XBF   = (unsigned short*)carve((size_t)N_ * C_ * 2);          // x bf16
  unsigned short* WQKV  = (unsigned short*)carve((size_t)3 * C_ * C_ * 2);      // qkv_w bf16
  unsigned short* WPROJ = (unsigned short*)carve((size_t)C_ * C_ * 2);          // proj_w bf16
  unsigned short* QBF   = (unsigned short*)carve((size_t)H_ * N_ * D_ * 2);     // (h,n,d)
  unsigned short* KBF   = (unsigned short*)carve((size_t)H_ * MPAD * D_ * 2);   // (h,m,d)
  unsigned short* VBFT  = (unsigned short*)carve((size_t)H_ * D_ * MPAD * 2);   // (h,d,m)
  unsigned short* ATTNBF= (unsigned short*)carve((size_t)N_ * C_ * 2);          // (n, h*64+d)
  float*          PART  = (float*)carve((size_t)H_ * 36 * PARTS * PSTRIDE * 4); // split-M partials
  (void)ws_size; (void)in_sizes; (void)n_in; (void)out_size;

  const int xN = N_ * C_, wqN = 3 * C_ * C_, wpN = C_ * C_;
  conv_bf16_kernel<<<(xN + 255) / 256, 256, 0, stream>>>(x, XBF, xN);
  conv_bf16_kernel<<<(wqN + 255) / 256, 256, 0, stream>>>(qkv_w, WQKV, wqN);
  conv_bf16_kernel<<<(wpN + 255) / 256, 256, 0, stream>>>(proj_w, WPROJ, wpN);

  const long kTot = (long)H_ * MPAD * 64;                 // 10,715,136 (exact multiple of 256)
  gather_k_kernel<<<(unsigned)(kTot / 256), 256, 0, stream>>>(k_cache, k_special, KBF);
  gather_vt_kernel<<<dim3(MPAD / 64, H_), 256, 0, stream>>>(v_cache, v_special, VBFT);

  qkv_gemm_kernel<<<dim3(N_ / 16, 3 * C_ / 64), 32, 0, stream>>>(XBF, WQKV, qkv_b, QBF, KBF, VBFT);
  attn_part_kernel<<<dim3(N_ / 16, H_, PARTS), 32, 0, stream>>>(QBF, KBF, VBFT, PART);
  attn_reduce_kernel<<<dim3(N_ / 16, H_), 128, 0, stream>>>(PART, ATTNBF);
  proj_gemm_kernel<<<dim3(N_ / 16, C_ / 64), 32, 0, stream>>>(ATTNBF, WPROJ, proj_b, out);
}
